// GraphGRU_15212774162623
// MI455X (gfx1250) — matrix operand order repaired and checked by reading, not compiled, not run on hardware
//
#include <hip/hip_runtime.h>
#include <hip/hip_bf16.h>
#include <math.h>

// Problem constants (match reference): B,S,N,DI,DH,NL = 8,128,512,64,64,2
#define GB  8
#define GS  128
#define GN  512
#define GDI 64
#define GDH 64

typedef unsigned short u16;
typedef unsigned int   u32;
typedef __attribute__((ext_vector_type(16))) __bf16 bf16x16;
typedef __attribute__((ext_vector_type(8)))  float  f32x8;

#define WMMA_BF16(a, b, c) \
    __builtin_amdgcn_wmma_f32_16x16x32_bf16(false, (a), false, (b), (short)0, (c), false, false)

__device__ __forceinline__ u16 f2bf(float f) {
    union { float f; u32 u; } v; v.f = f;
    u32 u = v.u;
    return (u16)((u + 0x7FFFu + ((u >> 16) & 1u)) >> 16);  // round-to-nearest-even
}

// ---- WMMA operand loaders -------------------------------------------------
// A-matrix 16x32 bf16 (ISA 7.12.2): lane holds row M=lane&15; elements 0-7 are
// K = kbase..kbase+7, elements 8-15 are K = kbase+16..kbase+23, kbase = (lane>=16)*8.
__device__ __forceinline__ bf16x16 load_a_tile(const u16* __restrict__ base,
                                               int r0, int ld, int kc, int lane) {
    const int r  = r0 + (lane & 15);
    const int kb = (lane >> 4) << 3;              // 0 or 8
    const u16* q = base + (size_t)r * ld + kc + kb;
    union { bf16x16 v; uint4 q4[2]; } u;
    u.q4[0] = *(const uint4*)(q);                 // K = kb .. kb+7
    u.q4[1] = *(const uint4*)(q + 16);            // K = kb+16 .. kb+23
    return u.v;
}

// B-matrix 32x16 bf16: stored K-major per column (we keep h / weights transposed):
// lane holds column N=lane&15; elements 0-7 are K = kb..kb+7, elements 8-15 are K = kb+16..kb+23, kb = (lane>=16)*8.
__device__ __forceinline__ bf16x16 load_b_tile(const u16* __restrict__ base,
                                               int n0, int ld, int kc, int lane) {
    const int nn = n0 + (lane & 15);
    const int kb = (lane >> 4) << 3;              // 0 or 8
    const u16* q = base + (size_t)nn * ld + kc + kb;
    union { bf16x16 v; uint4 q4[2]; } u;
    u.q4[0] = *(const uint4*)(q);
    u.q4[1] = *(const uint4*)(q + 16);
    return u.v;
}

__device__ __forceinline__ float sigmoidf_(float x) {
    return 1.0f / (1.0f + __expf(-x));
}

// ---- Pre-pass kernels -----------------------------------------------------
__global__ __launch_bounds__(256)
void softmax_row_bf16(const float* __restrict__ E, u16* __restrict__ out, int n) {
    __shared__ float red[256];
    const int row = blockIdx.x;
    const int tid = threadIdx.x;
    const float* er = E + (size_t)row * n;

    float mx = -3.4e38f;
    for (int i = tid; i < n; i += 256) mx = fmaxf(mx, er[i]);
    red[tid] = mx; __syncthreads();
    for (int s = 128; s > 0; s >>= 1) {
        if (tid < s) red[tid] = fmaxf(red[tid], red[tid + s]);
        __syncthreads();
    }
    mx = red[0]; __syncthreads();

    float sum = 0.f;
    for (int i = tid; i < n; i += 256) sum += __expf(er[i] - mx);
    red[tid] = sum; __syncthreads();
    for (int s = 128; s > 0; s >>= 1) {
        if (tid < s) red[tid] += red[tid + s];
        __syncthreads();
    }
    const float inv = 1.0f / red[0];
    for (int i = tid; i < n; i += 256)
        out[(size_t)row * n + i] = f2bf(__expf(er[i] - mx) * inv);
}

// W (K x Nn, row-major f32) -> out (Nn x K, row-major bf16)
__global__ __launch_bounds__(256)
void transpose_to_bf16(const float* __restrict__ W, u16* __restrict__ out, int K, int Nn) {
    const int idx = blockIdx.x * 256 + threadIdx.x;
    if (idx >= K * Nn) return;
    const int k = idx / Nn, nn = idx % Nn;
    out[(size_t)nn * K + k] = f2bf(W[idx]);
}

__global__ __launch_bounds__(256)
void zero_u4(uint4* __restrict__ p, int n4) {
    const int i = blockIdx.x * 256 + threadIdx.x;
    if (i < n4) { uint4 z; z.x = z.y = z.z = z.w = 0u; p[i] = z; }
}

__global__ __launch_bounds__(256)
void copy_f32(const float* __restrict__ a, float* __restrict__ b, int n) {
    const int i = blockIdx.x * 256 + threadIdx.x;
    if (i < n) b[i] = a[i];
}

// ---- Fused GRU step -------------------------------------------------------
// grid = B * (N/16) blocks, 128 threads (4 waves). Wave w owns D-cols [16w,16w+16).
__global__ __launch_bounds__(128)
void gru_step(const float* __restrict__ xseq, int t,
              const u16*  __restrict__ Esm,                     // N x N bf16 (row-major)
              const u16*  __restrict__ WgT,                     // 2DH x (DI+DH) bf16 (K-major)
              const float* __restrict__ bg,                     // 2DH
              const u16*  __restrict__ WcT,                     // DH x (DI+DH) bf16 (K-major)
              const float* __restrict__ bc,                     // DH
              float* __restrict__ hF,                           // B*N*DH f32 (current h)
              const u16* __restrict__ hTin,                     // B*DH*N bf16 (prev h, transposed)
              u16*  __restrict__ hTout,                         // B*DH*N bf16 (next h, transposed)
              float* __restrict__ yseq)                         // B*S*N*DH f32 output
{
    __shared__ __align__(16) u16  sA[16 * 128];   // [x_t | wh]  then  [x_t | r*wh]
    __shared__ float sWh[16 * GDH];
    __shared__ float sZ [16 * GDH];

    const int tid   = threadIdx.x;
    const int lane  = tid & 31;
    const int w     = tid >> 5;
    const int b     = blockIdx.x >> 5;             // N/16 == 32 tiles
    const int j0    = (blockIdx.x & 31) * 16;
    const int n     = lane & 15;                   // column within 16-wide tile
    const int mbase = (lane >> 4) << 3;            // acc row base (0 or 8)

    // ---- Phase 1: wh tile = Esm[j0:j0+16, :] @ h[b]  (this wave: cols 16w..16w+15)
    f32x8 acc = {0.f, 0.f, 0.f, 0.f, 0.f, 0.f, 0.f, 0.f};
    const u16* hTb = hTin + (size_t)b * GDH * GN;
    for (int kc = 0; kc < GN; kc += 32) {
        if (kc + 32 < GN)
            __builtin_prefetch(Esm + (size_t)(j0 + n) * GN + kc + 32, 0, 0);
        bf16x16 a  = load_a_tile(Esm, j0,     GN, kc, lane);
        bf16x16 bm = load_b_tile(hTb, w * 16, GN, kc, lane);
        acc = WMMA_BF16(a, bm, acc);
    }
    // stash wh (f32 for r*wh, bf16 into concat tile cols 64..127)
    {
        const int d = w * 16 + n;
        #pragma unroll
        for (int v = 0; v < 8; ++v) {
            const int m = mbase + v;
            const float val = acc[v];
            sWh[m * GDH + d]      = val;
            sA [m * 128 + 64 + d] = f2bf(val);
        }
    }
    // x_t tile -> sA cols 0..63 (16x64, bf16); all 128 threads cooperate
    for (int i = tid; i < 16 * 64; i += 128) {
        const int m = i >> 6, dd = i & 63;
        const float xv = xseq[(((size_t)b * GS + t) * GN + j0 + m) * GDI + dd];
        sA[m * 128 + dd] = f2bf(xv);
    }
    __syncthreads();

    // ---- Phase 2: gates = sigmoid([x|wh] @ Wg + bg); wave w: cols [32w, 32w+32)
    f32x8 g0 = {0.f,0.f,0.f,0.f,0.f,0.f,0.f,0.f};
    f32x8 g1 = {0.f,0.f,0.f,0.f,0.f,0.f,0.f,0.f};
    for (int kc = 0; kc < 128; kc += 32) {
        bf16x16 a  = load_a_tile(sA, 0, 128, kc, lane);
        bf16x16 b0 = load_b_tile(WgT, w * 32,      128, kc, lane);
        bf16x16 b1 = load_b_tile(WgT, w * 32 + 16, 128, kc, lane);
        g0 = WMMA_BF16(a, b0, g0);
        g1 = WMMA_BF16(a, b1, g1);
    }
    const int c0 = w * 32 + n;
    const int c1 = w * 32 + 16 + n;
    {
        const float bb0 = bg[c0], bb1 = bg[c1];
        #pragma unroll
        for (int v = 0; v < 8; ++v) {
            g0[v] = sigmoidf_(g0[v] + bb0);
            g1[v] = sigmoidf_(g1[v] + bb1);
        }
    }
    __syncthreads();   // everyone done reading sA before r*wh overwrite

    if (w < 2) {       // waves 0,1 hold r (gate cols 0..63): write r*wh into sA cols 64..127
        #pragma unroll
        for (int v = 0; v < 8; ++v) {
            const int m = mbase + v;
            sA[m * 128 + 64 + c0] = f2bf(g0[v] * sWh[m * GDH + c0]);
            sA[m * 128 + 64 + c1] = f2bf(g1[v] * sWh[m * GDH + c1]);
        }
    } else {           // waves 2,3 hold z (gate cols 64..127): stash to LDS
        #pragma unroll
        for (int v = 0; v < 8; ++v) {
            const int m = mbase + v;
            sZ[m * GDH + (c0 - 64)] = g0[v];
            sZ[m * GDH + (c1 - 64)] = g1[v];
        }
    }
    __syncthreads();

    // ---- Phase 3: can = tanh([x | r*wh] @ Wc + bc); wave w: cols [16w, 16w+16)
    f32x8 cacc = {0.f,0.f,0.f,0.f,0.f,0.f,0.f,0.f};
    for (int kc = 0; kc < 128; kc += 32) {
        bf16x16 a  = load_a_tile(sA, 0, 128, kc, lane);
        bf16x16 bw = load_b_tile(WcT, w * 16, 128, kc, lane);
        cacc = WMMA_BF16(a, bw, cacc);
    }

    // ---- Phase 4: h' = (1-z)*h + z*can ; write f32 state, bf16ᵀ state, y_t
    {
        const int d = w * 16 + n;
        const float bcv = bc[d];
        float* hFb  = hF   + (size_t)b * GN * GDH;
        u16*   hTo  = hTout + (size_t)b * GDH * GN;
        float* yb   = yseq + (((size_t)b * GS + t) * GN) * GDH;
        #pragma unroll
        for (int v = 0; v < 8; ++v) {
            const int m = mbase + v;
            const int j = j0 + m;
            const float can  = tanhf(cacc[v] + bcv);
            const float z    = sZ[m * GDH + d];
            const float hold = hFb[(size_t)j * GDH + d];
            const float hnew = (1.0f - z) * hold + z * can;
            hFb[(size_t)j * GDH + d] = hnew;
            hTo[(size_t)d * GN + j]  = f2bf(hnew);
            yb [(size_t)j * GDH + d] = hnew;
        }
    }
}

// ---- Host driver ----------------------------------------------------------
extern "C" void kernel_launch(void* const* d_in, const int* in_sizes, int n_in,
                              void* d_out, int out_size, void* d_ws, size_t ws_size,
                              hipStream_t stream) {
    (void)in_sizes; (void)n_in; (void)out_size; (void)ws_size;

    const float* x   = (const float*)d_in[0];
    const float* E0  = (const float*)d_in[1];
    const float* Wg0 = (const float*)d_in[2];
    const float* bg0 = (const float*)d_in[3];
    const float* Wc0 = (const float*)d_in[4];
    const float* bc0 = (const float*)d_in[5];
    const float* E1  = (const float*)d_in[6];
    const float* Wg1 = (const float*)d_in[7];
    const float* bg1 = (const float*)d_in[8];
    const float* Wc1 = (const float*)d_in[9];
    const float* bc1 = (const float*)d_in[10];

    char* ws = (char*)d_ws;
    size_t off = 0;
    auto walloc = [&](size_t bytes) -> void* {
        void* p = ws + off;
        off = (off + bytes + 255) & ~(size_t)255;
        return p;
    };

    u16*   esm0 = (u16*)  walloc((size_t)GN * GN * 2);
    u16*   esm1 = (u16*)  walloc((size_t)GN * GN * 2);
    u16*   wgT0 = (u16*)  walloc((size_t)128 * 128 * 2);
    u16*   wcT0 = (u16*)  walloc((size_t)64  * 128 * 2);
    u16*   wgT1 = (u16*)  walloc((size_t)128 * 128 * 2);
    u16*   wcT1 = (u16*)  walloc((size_t)64  * 128 * 2);
    float* hF   = (float*)walloc((size_t)GB * GN * GDH * 4);
    u16*   hT0  = (u16*)  walloc((size_t)GB * GDH * GN * 2);
    u16*   hT1  = (u16*)  walloc((size_t)GB * GDH * GN * 2);
    float* y0   = (float*)walloc((size_t)GB * GS * GN * GDH * 4);

    float* cur  = (float*)d_out;
    float* last = (float*)d_out + (size_t)GB * GS * GN * GDH;

    // Pre-pass: softmax(E) -> bf16, weights -> bf16 transposed (K-major per out col)
    softmax_row_bf16<<<GN, 256, 0, stream>>>(E0, esm0, GN);
    softmax_row_bf16<<<GN, 256, 0, stream>>>(E1, esm1, GN);
    transpose_to_bf16<<<(128 * 128 + 255) / 256, 256, 0, stream>>>(Wg0, wgT0, 128, 128);
    transpose_to_bf16<<<(128 * 64  + 255) / 256, 256, 0, stream>>>(Wc0, wcT0, 128, 64);
    transpose_to_bf16<<<(128 * 128 + 255) / 256, 256, 0, stream>>>(Wg1, wgT1, 128, 128);
    transpose_to_bf16<<<(128 * 64  + 255) / 256, 256, 0, stream>>>(Wc1, wcT1, 128, 64);

    auto zero = [&](void* p, size_t bytes) {
        const int n4 = (int)(bytes / 16);
        zero_u4<<<(n4 + 255) / 256, 256, 0, stream>>>((uint4*)p, n4);
    };

    auto run_layer = [&](const float* xin, const u16* esm,
                         const u16* wgT, const float* bgp,
                         const u16* wcT, const float* bcp,
                         float* yout, float* lastout) {
        zero(hF,  (size_t)GB * GN * GDH * 4);
        zero(hT0, (size_t)GB * GDH * GN * 2);
        u16* hp = hT0;
        u16* hn = hT1;
        for (int t = 0; t < GS; ++t) {
            gru_step<<<GB * (GN / 16), 128, 0, stream>>>(
                xin, t, esm, wgT, bgp, wcT, bcp, hF, hp, hn, yout);
            u16* tmp = hp; hp = hn; hn = tmp;
        }
        const int nh = GB * GN * GDH;
        copy_f32<<<(nh + 255) / 256, 256, 0, stream>>>(hF, lastout, nh);
    };

    run_layer(x,  esm0, wgT0, bg0, wcT0, bc0, y0,  last);
    run_layer(y0, esm1, wgT1, bg1, wcT1, bc1, cur, last + (size_t)GB * GN * GDH);
}
